// WindowAttention_73212012528186
// MI455X (gfx1250) — matrix operand order repaired and checked by reading, not compiled, not run on hardware
//
#include <hip/hip_runtime.h>

typedef __attribute__((ext_vector_type(16))) _Float16 v16h;
typedef __attribute__((ext_vector_type(8)))  float    v8f;
typedef __attribute__((ext_vector_type(4)))  float    f4;
typedef __attribute__((ext_vector_type(4)))  unsigned int u4x;
typedef __attribute__((ext_vector_type(2)))  unsigned int u2x;

#define NTOK   49
#define HEADS  6
#define CDIM   192
#define NQKV   576
#define BWIN   8192
#define MTOT   (BWIN * NTOK)     // 401408 = 128 * 3136

union FragH { v16h h; f4 f[2]; };
union Acc8  { v8f v; float f[8]; };

// ---------------------------------------------------------------------------
// CDNA5 async global->LDS copy (16 bytes per lane), tracked by ASYNCcnt.
// VDST = per-lane LDS byte address (low 32 bits of generic pointer),
// VADDR = 64-bit global address. ISA 15.18.3 opcode 98.
// ---------------------------------------------------------------------------
__device__ __forceinline__ void async_copy16(void* ldsDst, const void* gSrc) {
  unsigned lds = (unsigned)(unsigned long long)(uintptr_t)ldsDst;
  unsigned long long g = (unsigned long long)(uintptr_t)gSrc;
  asm volatile("global_load_async_to_lds_b128 %0, %1, off"
               :: "v"(lds), "v"(g) : "memory");
}
__device__ __forceinline__ void async_wait0() {
  asm volatile("s_wait_asynccnt 0x0" ::: "memory");
}

// ---------------------------------------------------------------------------
// Kernel 1: weight fp32->f16 conversion + relative-position bias matrix build
// ---------------------------------------------------------------------------
__global__ void prep_kernel(const float* __restrict__ qkv_w,
                            const float* __restrict__ proj_w,
                            const float* __restrict__ bias_table,
                            _Float16* __restrict__ qkvw_h,
                            _Float16* __restrict__ projw_h,
                            float* __restrict__ biasM) {
  const int total = CDIM * NQKV + CDIM * CDIM + HEADS * NTOK * NTOK;
  for (int i = blockIdx.x * blockDim.x + threadIdx.x; i < total;
       i += gridDim.x * blockDim.x) {
    if (i < CDIM * NQKV) {
      qkvw_h[i] = (_Float16)qkv_w[i];
    } else if (i < CDIM * NQKV + CDIM * CDIM) {
      int j = i - CDIM * NQKV;
      projw_h[j] = (_Float16)proj_w[j];
    } else {
      int j = i - CDIM * NQKV - CDIM * CDIM;
      int h = j / (NTOK * NTOK);
      int rem = j % (NTOK * NTOK);
      int ti = rem / NTOK, tj = rem % NTOK;
      int d0 = (ti % 7) - (tj % 7) + 6;   // x delta + (ws-1)
      int d1 = (ti / 7) - (tj / 7) + 6;   // y delta + (ws-1)
      int idx = d0 * 13 + d1;             // * (2*ws-1)
      biasM[h * NTOK * NTOK + rem] = bias_table[idx * HEADS + h];
    }
  }
}

// ---------------------------------------------------------------------------
// Kernel 2: QKV projection GEMM  (M x 192) x (192 x 576) + bias -> f16
// Block tile 128x64, 8 waves, each wave 32x32 (2x2 WMMA tiles).
// ---------------------------------------------------------------------------
__global__ __launch_bounds__(256)
void qkv_gemm_kernel(const float* __restrict__ X,      // M x 192 fp32
                     const _Float16* __restrict__ Wh,  // 192 x 576 f16
                     const float* __restrict__ bvec,   // 576
                     _Float16* __restrict__ Out)       // M x 576 f16
{
  __shared__ _Float16 sA[128][40];   // [m][k], pad 8 halves (16B) keeps b128 align
  __shared__ _Float16 sB[64][40];    // [n][k] (transposed)
  const int tid  = threadIdx.x;
  const int lane = tid & 31;
  const int wv   = tid >> 5;
  const int wm   = wv & 3;           // wave row 0..3  (32 rows each)
  const int wn   = wv >> 2;          // wave col 0..1  (32 cols each)
  const int mBase = blockIdx.x * 128;
  const int nBase = blockIdx.y * 64;
  const int ls = lane & 15, hi = lane >> 4;

  v8f acc[2][2] = {};

  for (int k0 = 0; k0 < CDIM; k0 += 32) {
    __syncthreads();
    // --- A tile: 128x32 fp32 -> f16 into LDS ---
#pragma unroll
    for (int it = 0; it < 4; ++it) {
      int c = tid + it * 256;          // 0..1023
      int row = c >> 3, seg = c & 7;   // seg: 4-float chunk
      const float* gp = X + (size_t)(mBase + row) * CDIM + k0 + seg * 4;
      f4 v = *(const f4*)gp;
      if (k0 + 32 < CDIM) __builtin_prefetch(gp + 32, 0, 1);
      union { _Float16 h[4]; u2x u; } pk;
      pk.h[0] = (_Float16)v.x; pk.h[1] = (_Float16)v.y;
      pk.h[2] = (_Float16)v.z; pk.h[3] = (_Float16)v.w;
      *(u2x*)&sA[row][seg * 4] = pk.u;
    }
    // --- B tile: 32x64 f16, stored transposed [n][k] ---
    {
      int kk = tid >> 3, nseg = tid & 7;
      union { u4x u; _Float16 h[8]; } ld;
      ld.u = *(const u4x*)(Wh + (size_t)(k0 + kk) * NQKV + nBase + nseg * 8);
#pragma unroll
      for (int e = 0; e < 8; ++e) sB[nseg * 8 + e][kk] = ld.h[e];
    }
    __syncthreads();

    FragH a[2], b[2];
#pragma unroll
    for (int im = 0; im < 2; ++im) {
      int m = wm * 32 + im * 16 + ls;
      a[im].f[0] = *(const f4*)&sA[m][hi * 8];        // K = hi*8 .. +7
      a[im].f[1] = *(const f4*)&sA[m][16 + hi * 8];   // K = 16+hi*8 .. +7
    }
#pragma unroll
    for (int in = 0; in < 2; ++in) {
      int n = wn * 32 + in * 16 + ls;
      b[in].f[0] = *(const f4*)&sB[n][hi * 8];        // K = hi*8 .. +7
      b[in].f[1] = *(const f4*)&sB[n][16 + hi * 8];   // K = 16+hi*8 .. +7
    }
#pragma unroll
    for (int im = 0; im < 2; ++im)
#pragma unroll
      for (int in = 0; in < 2; ++in)
        acc[im][in] = __builtin_amdgcn_wmma_f32_16x16x32_f16(
            false, a[im].h, false, b[in].h, (short)0, acc[im][in], false, false);
  }

  // --- store: bias add, fp32 -> f16 ---
#pragma unroll
  for (int in = 0; in < 2; ++in) {
    int col = nBase + wn * 32 + in * 16 + ls;
    float bq = bvec[col];
#pragma unroll
    for (int im = 0; im < 2; ++im) {
      Acc8 u; u.v = acc[im][in];
      int rb = mBase + wm * 32 + im * 16 + hi * 8;
#pragma unroll
      for (int r = 0; r < 8; ++r)
        Out[(size_t)(rb + r) * NQKV + col] = (_Float16)(u.f[r] + bq);
    }
  }
}

// ---------------------------------------------------------------------------
// Kernel 3: per-(window, head) attention. 4 waves, 49x49 padded to 64x64.
// Q/K staged with CDNA5 async-to-LDS; V transposed manually.
// ---------------------------------------------------------------------------
__global__ __launch_bounds__(128)
void attn_kernel(const _Float16* __restrict__ QKV,  // M x 576 f16
                 const float* __restrict__ biasM,   // 6 x 49 x 49
                 _Float16* __restrict__ AOut)       // M x 192 f16
{
  __shared__ _Float16 sQ[64][40];
  __shared__ _Float16 sK[64][40];
  __shared__ _Float16 sVT[32][72];   // [d][token]
  __shared__ _Float16 sP[64][72];    // softmaxed probabilities
  __shared__ float    sBias[NTOK * NTOK];

  const int tid  = threadIdx.x;
  const int lane = tid & 31;
  const int wv   = tid >> 5;
  const int w    = blockIdx.x / HEADS;
  const int h    = blockIdx.x % HEADS;
  const size_t rowbase = (size_t)w * NTOK;
  const int ls = lane & 15, hi = lane >> 4;
  const int mrow = wv * 16;

  // zero-fill Q/K/V staging (padding rows/cols must be exactly 0.0)
  {
    unsigned int* zq = (unsigned int*)&sQ[0][0];
    unsigned int* zk = (unsigned int*)&sK[0][0];
    for (int i = tid; i < 64 * 40 / 2; i += 128) { zq[i] = 0u; zk[i] = 0u; }
    unsigned int* zv = (unsigned int*)&sVT[0][0];
    for (int i = tid; i < 32 * 72 / 2; i += 128) zv[i] = 0u;
  }
  __syncthreads();

  // load Q, K via async global->LDS (ASYNCcnt) and V (transposed scatter)
  for (int c = tid; c < NTOK * 4; c += 128) {
    int token = c >> 2, dseg = c & 3;
    const _Float16* base = QKV + (rowbase + token) * NQKV + h * 32 + dseg * 8;
    async_copy16(&sQ[token][dseg * 8], base);
    async_copy16(&sK[token][dseg * 8], base + CDIM);
    union { u4x u; _Float16 hh[8]; } vv;
    vv.u = *(const u4x*)(base + 2 * CDIM);
#pragma unroll
    for (int e = 0; e < 8; ++e) sVT[dseg * 8 + e][token] = vv.hh[e];
  }
  for (int i = tid; i < NTOK * NTOK; i += 128)
    sBias[i] = biasM[h * NTOK * NTOK + i];
  async_wait0();
  __syncthreads();

  // --- S = Q * K^T : wave wv owns rows [mrow, mrow+16), 4 WMMAs across cols ---
  FragH qa;
  {
    int m = mrow + ls;
    qa.f[0] = *(const f4*)&sQ[m][hi * 8];
    qa.f[1] = *(const f4*)&sQ[m][16 + hi * 8];
  }
  float S[4][8];
#pragma unroll
  for (int jt = 0; jt < 4; ++jt) {
    FragH kb;
    int n = jt * 16 + ls;                 // column token j
    kb.f[0] = *(const f4*)&sK[n][hi * 8];
    kb.f[1] = *(const f4*)&sK[n][16 + hi * 8];
    v8f z = {};
    v8f s = __builtin_amdgcn_wmma_f32_16x16x32_f16(
        false, qa.h, false, kb.h, (short)0, z, false, false);
    Acc8 u; u.v = s;
#pragma unroll
    for (int r = 0; r < 8; ++r) S[jt][r] = u.f[r];
  }

  // --- softmax over columns, per row, using C-layout + shfl_xor(width 16) ---
  const float scale = 0.17677669529663687f;  // 32^-0.5
#pragma unroll
  for (int r = 0; r < 8; ++r) {
    int row = mrow + hi * 8 + r;
    float vals[4];
#pragma unroll
    for (int jt = 0; jt < 4; ++jt) {
      int col = jt * 16 + ls;
      float sv = S[jt][r] * scale;
      bool ok = (row < NTOK) && (col < NTOK);
      vals[jt] = ok ? (sv + sBias[row * NTOK + col]) : -3.0e38f;
    }
    float mx = fmaxf(fmaxf(vals[0], vals[1]), fmaxf(vals[2], vals[3]));
#pragma unroll
    for (int off = 8; off >= 1; off >>= 1)
      mx = fmaxf(mx, __shfl_xor(mx, off, 16));
    float p[4];
    float ssum = 0.f;
#pragma unroll
    for (int jt = 0; jt < 4; ++jt) {
      float e = (vals[jt] <= -1.0e38f) ? 0.f : __expf(vals[jt] - mx);
      p[jt] = e; ssum += e;
    }
#pragma unroll
    for (int off = 8; off >= 1; off >>= 1)
      ssum += __shfl_xor(ssum, off, 16);
    float inv = (ssum > 0.f) ? (1.f / ssum) : 0.f;
#pragma unroll
    for (int jt = 0; jt < 4; ++jt)
      sP[row][jt * 16 + ls] = (_Float16)(p[jt] * inv);
  }

  // --- O = P * V : K=64 in two 32-steps, 2 column tiles of hd ---
  v8f oacc[2] = {};
#pragma unroll
  for (int kt = 0; kt < 2; ++kt) {
    FragH pa;
    int m = mrow + ls;
    pa.f[0] = *(const f4*)&sP[m][kt * 32 + hi * 8];
    pa.f[1] = *(const f4*)&sP[m][kt * 32 + 16 + hi * 8];
#pragma unroll
    for (int nt = 0; nt < 2; ++nt) {
      FragH vb;
      int n = nt * 16 + ls;               // d index
      vb.f[0] = *(const f4*)&sVT[n][kt * 32 + hi * 8];
      vb.f[1] = *(const f4*)&sVT[n][kt * 32 + 16 + hi * 8];
      oacc[nt] = __builtin_amdgcn_wmma_f32_16x16x32_f16(
          false, pa.h, false, vb.h, (short)0, oacc[nt], false, false);
    }
  }

#pragma unroll
  for (int nt = 0; nt < 2; ++nt) {
    Acc8 u; u.v = oacc[nt];
#pragma unroll
    for (int r = 0; r < 8; ++r) {
      int row = mrow + hi * 8 + r;
      if (row < NTOK) {
        int col = nt * 16 + ls;
        AOut[(rowbase + row) * CDIM + h * 32 + col] = (_Float16)u.f[r];
      }
    }
  }
}

// ---------------------------------------------------------------------------
// Kernel 4: output projection GEMM (M x 192) x (192 x 192) + bias -> fp32 out
// A tile staged with CDNA5 async global->LDS (pure f16 copy, no transform).
// ---------------------------------------------------------------------------
__global__ __launch_bounds__(256)
void proj_gemm_kernel(const _Float16* __restrict__ Ah,  // M x 192 f16
                      const _Float16* __restrict__ Wh,  // 192 x 192 f16
                      const float* __restrict__ bvec,   // 192
                      float* __restrict__ Out)          // M x 192 fp32
{
  __shared__ _Float16 sA[128][40];
  __shared__ _Float16 sB[64][40];
  const int tid  = threadIdx.x;
  const int lane = tid & 31;
  const int wv   = tid >> 5;
  const int wm   = wv & 3;
  const int wn   = wv >> 2;
  const int mBase = blockIdx.x * 128;
  const int nBase = blockIdx.y * 64;
  const int ls = lane & 15, hi = lane >> 4;

  v8f acc[2][2] = {};

  for (int k0 = 0; k0 < CDIM; k0 += 32) {
    __syncthreads();
    // --- A tile: 128x32 f16, async global->LDS, 2 x 16B per thread ---
#pragma unroll
    for (int it = 0; it < 2; ++it) {
      int c = tid + it * 256;            // 0..511
      int row = c >> 2, seg = c & 3;     // seg: 8-half chunk
      async_copy16(&sA[row][seg * 8],
                   Ah + (size_t)(mBase + row) * CDIM + k0 + seg * 8);
    }
    // --- B tile: 32x64 f16, stored transposed [n][k] ---
    {
      int kk = tid >> 3, nseg = tid & 7;
      union { u4x u; _Float16 h[8]; } ld;
      ld.u = *(const u4x*)(Wh + (size_t)(k0 + kk) * CDIM + nBase + nseg * 8);
#pragma unroll
      for (int e = 0; e < 8; ++e) sB[nseg * 8 + e][kk] = ld.h[e];
    }
    async_wait0();
    __syncthreads();

    FragH a[2], b[2];
#pragma unroll
    for (int im = 0; im < 2; ++im) {
      int m = wm * 32 + im * 16 + ls;
      a[im].f[0] = *(const f4*)&sA[m][hi * 8];
      a[im].f[1] = *(const f4*)&sA[m][16 + hi * 8];
    }
#pragma unroll
    for (int in = 0; in < 2; ++in) {
      int n = wn * 32 + in * 16 + ls;
      b[in].f[0] = *(const f4*)&sB[n][hi * 8];
      b[in].f[1] = *(const f4*)&sB[n][16 + hi * 8];
    }
#pragma unroll
    for (int im = 0; im < 2; ++im)
#pragma unroll
      for (int in = 0; in < 2; ++in)
        acc[im][in] = __builtin_amdgcn_wmma_f32_16x16x32_f16(
            false, a[im].h, false, b[in].h, (short)0, acc[im][in], false, false);
  }

#pragma unroll
  for (int in = 0; in < 2; ++in) {
    int col = nBase + wn * 32 + in * 16 + ls;
    float bq = bvec[col];
#pragma unroll
    for (int im = 0; im < 2; ++im) {
      Acc8 u; u.v = acc[im][in];
      int rb = mBase + wm * 32 + im * 16 + hi * 8;
#pragma unroll
      for (int r = 0; r < 8; ++r)
        Out[(size_t)(rb + r) * CDIM + col] = u.f[r] + bq;
    }
  }
}

// ---------------------------------------------------------------------------
extern "C" void kernel_launch(void* const* d_in, const int* in_sizes, int n_in,
                              void* d_out, int out_size, void* d_ws, size_t ws_size,
                              hipStream_t stream) {
  const float* x          = (const float*)d_in[0];
  const float* qkv_w      = (const float*)d_in[1];
  const float* qkv_b      = (const float*)d_in[2];
  const float* proj_w     = (const float*)d_in[3];
  const float* proj_b     = (const float*)d_in[4];
  const float* bias_table = (const float*)d_in[5];
  float* out = (float*)d_out;

  char* ws = (char*)d_ws;
  size_t off = 0;
  _Float16* qkvw_h  = (_Float16*)(ws + off); off += (size_t)CDIM * NQKV * 2;
  _Float16* projw_h = (_Float16*)(ws + off); off += (size_t)CDIM * CDIM * 2;
  float*    biasM   = (float*)(ws + off);    off += (size_t)HEADS * NTOK * NTOK * 4;
  off = (off + 255) & ~(size_t)255;
  _Float16* qkvbuf  = (_Float16*)(ws + off); off += (size_t)MTOT * NQKV * 2;
  _Float16* aout    = (_Float16*)(ws + off); off += (size_t)MTOT * CDIM * 2;

  prep_kernel<<<640, 256, 0, stream>>>(qkv_w, proj_w, bias_table,
                                       qkvw_h, projw_h, biasM);
  qkv_gemm_kernel<<<dim3(MTOT / 128, NQKV / 64), 256, 0, stream>>>(
      x, qkvw_h, qkv_b, qkvbuf);
  attn_kernel<<<dim3(BWIN * HEADS), 128, 0, stream>>>(qkvbuf, biasM, aout);
  proj_gemm_kernel<<<dim3(MTOT / 128, CDIM / 64), 256, 0, stream>>>(
      aout, projw_h, proj_b, out);
}
